// LlamaAttention_28501402976473
// MI455X (gfx1250) — compile-verified, matrix operand order checked by reading, not run on hardware
//
#include <hip/hip_runtime.h>
#include <math.h>

typedef unsigned short u16;
typedef __attribute__((ext_vector_type(16))) __bf16 bf16x16;
typedef __attribute__((ext_vector_type(8)))  float  f32x8;

union FragB {
    bf16x16 v;
    uint4   u[2];
};

union V16 {
    uint4 q[2];
    u16   h[16];
};

__device__ __forceinline__ u16 f2bf(float f) {
    unsigned int u = __float_as_uint(f);
    unsigned int r = u + 0x7FFFu + ((u >> 16) & 1u);
    return (u16)(r >> 16);
}

__device__ __forceinline__ f32x8 wmma_bf16(const FragB& a, const FragB& b, f32x8 c) {
    return __builtin_amdgcn_wmma_f32_16x16x32_bf16(
        /*neg_a=*/false, a.v, /*neg_b=*/false, b.v,
        /*c_mod=*/(short)0, c, /*reuse_a=*/false, /*reuse_b=*/false);
}

// LDS byte address of a __shared__ object (generic pointer low 32 bits = LDS offset)
__device__ __forceinline__ unsigned lds_addr(const void* p) {
    return (unsigned)(unsigned long long)p;
}

// Async copy of 64B (global -> LDS), per-lane addressing; tracked by ASYNCcnt.
__device__ __forceinline__ void async_copy64(unsigned la, unsigned long long ga) {
    asm volatile(
        "global_load_async_to_lds_b128 %0, %1, off\n\t"
        "global_load_async_to_lds_b128 %0, %1, off offset:16\n\t"
        "global_load_async_to_lds_b128 %0, %1, off offset:32\n\t"
        "global_load_async_to_lds_b128 %0, %1, off offset:48"
        :: "v"(la), "v"(ga) : "memory");
}

__device__ __forceinline__ void async_copy32(unsigned la, unsigned long long ga) {
    asm volatile(
        "global_load_async_to_lds_b128 %0, %1, off\n\t"
        "global_load_async_to_lds_b128 %0, %1, off offset:16"
        :: "v"(la), "v"(ga) : "memory");
}

__device__ __forceinline__ void wait_async0() {
    asm volatile("s_wait_asynccnt 0x0" ::: "memory");
}

#define B_C   2
#define S_C   2048
#define H_C   4096
#define NH_C  32
#define NKV_C 8
#define HD_C  128

// ---------------------------------------------------------------------------
// f32 -> bf16 cast
// ---------------------------------------------------------------------------
__global__ __launch_bounds__(256)
void cast_bf16_kernel(const float* __restrict__ in, u16* __restrict__ out, long n) {
    long i = (long)blockIdx.x * blockDim.x + threadIdx.x;
    if (i < n) out[i] = f2bf(in[i]);
}

// ---------------------------------------------------------------------------
// Interleaved RoPE + cast to bf16.  Layout [B, S, nh, HD], one thread / pair.
// ---------------------------------------------------------------------------
__global__ __launch_bounds__(256)
void rope_cast_kernel(const float* __restrict__ in, const int* __restrict__ pos,
                      u16* __restrict__ out, int nheads, long total_pairs) {
    long i = (long)blockIdx.x * blockDim.x + threadIdx.x;
    if (i >= total_pairs) return;
    int  p    = (int)(i & 63);               // HD/2 = 64 pairs
    long rest = i >> 6;
    long bs   = rest / nheads;               // b*S + s
    int  s    = (int)(bs & (S_C - 1));
    int  b    = (int)(bs >> 11);             // /2048
    float ang = (float)pos[(size_t)b * S_C + s] * __powf(1.0e6f, -(float)p / 64.0f);
    float c, sn;
    sincosf(ang, &sn, &c);
    float x0 = in[2 * i], x1 = in[2 * i + 1];
    out[2 * i]     = f2bf(x0 * c - x1 * sn);
    out[2 * i + 1] = f2bf(x1 * c + x0 * sn);
}

// ---------------------------------------------------------------------------
// bf16 GEMM: C[M,N] = A[M,K] * W[N,K]^T + bias,  f32 output.
// 256 threads = 8 waves; tile 128x256; wave tile 64x64 (4x4 WMMA accums).
// Double-buffered LDS filled with async global->LDS copies (ASYNCcnt).
// ---------------------------------------------------------------------------
__global__ __launch_bounds__(256)
void gemm_bf16_kernel(const u16* __restrict__ A, const u16* __restrict__ W,
                      const float* __restrict__ bias, float* __restrict__ C,
                      int M, int N, int K) {
    __shared__ u16 lA[2][128][40];   // 20 KB : 32 used + 8 pad halfwords
    __shared__ u16 lB[2][256][40];   // 40 KB

    const int tid  = threadIdx.x;
    const int lane = tid & 31;
    const int w    = tid >> 5;
    const int wm   = (w >> 2) * 64;      // wave M offset in tile
    const int wn   = (w & 3) * 64;       // wave N offset in tile
    const int r    = lane & 15;
    const int half = lane >> 4;
    const int kb   = half * 8;

    const int bm = blockIdx.x * 128;
    const int bn = blockIdx.y * 256;

    // staging assignment: A: thread -> (row tid>>1, 32B strip); B: row tid, 64B strip
    const size_t gaA = (size_t)(bm + (tid >> 1)) * K + (tid & 1) * 16;
    const size_t gaB = (size_t)(bn + tid) * K;
    const unsigned laA0 = lds_addr(&lA[0][tid >> 1][(tid & 1) * 16]);
    const unsigned laA1 = lds_addr(&lA[1][tid >> 1][(tid & 1) * 16]);
    const unsigned laB0 = lds_addr(&lB[0][tid][0]);
    const unsigned laB1 = lds_addr(&lB[1][tid][0]);

    f32x8 acc[4][4] = {};

    const int nk = K >> 5;
    // prefetch k-step 0 into buffer 0
    async_copy32(laA0, (unsigned long long)(A + gaA));
    async_copy64(laB0, (unsigned long long)(W + gaB));

    for (int ks = 0; ks < nk; ++ks) {
        const int cur = ks & 1;
        wait_async0();          // my fills of buf[cur] complete
        __syncthreads();        // everyone's fills complete; buf[cur^1] free
        if (ks + 1 < nk) {
            const int k0n = (ks + 1) << 5;
            async_copy32(cur ? laA0 : laA1, (unsigned long long)(A + gaA + k0n));
            async_copy64(cur ? laB0 : laB1, (unsigned long long)(W + gaB + k0n));
        }

        FragB af[4];
        #pragma unroll
        for (int mi = 0; mi < 4; ++mi) {
            const u16* p = &lA[cur][wm + mi * 16 + r][kb];
            af[mi].u[0] = *(const uint4*)p;
            af[mi].u[1] = *(const uint4*)(p + 16);
        }
        #pragma unroll
        for (int ni = 0; ni < 4; ++ni) {
            FragB bf;
            const u16* p = &lB[cur][wn + ni * 16 + r][kb];
            bf.u[0] = *(const uint4*)p;
            bf.u[1] = *(const uint4*)(p + 16);
            #pragma unroll
            for (int mi = 0; mi < 4; ++mi)
                acc[mi][ni] = wmma_bf16(af[mi], bf, acc[mi][ni]);
        }
    }

    #pragma unroll
    for (int mi = 0; mi < 4; ++mi) {
        #pragma unroll
        for (int ni = 0; ni < 4; ++ni) {
            int col = bn + wn + ni * 16 + r;
            float bv = bias[col];
            #pragma unroll
            for (int v = 0; v < 8; ++v) {
                int row = bm + wm + mi * 16 + 8 * half + v;
                C[(size_t)row * N + col] = acc[mi][ni][v] + bv;
            }
        }
    }
}

// ---------------------------------------------------------------------------
// Flash attention (causal, GQA 4:1).  Block = 4 waves covering a 64-row
// q-block of one (b, h); each wave owns a 16-row tile, full 16x128 f32 acc.
// Keys in blocks of 32: K staged row-major in LDS via async copies,
// V staged transposed with paired 32-bit stores.
// ---------------------------------------------------------------------------
__global__ __launch_bounds__(128)
void attn_kernel(const u16* __restrict__ Qb, const u16* __restrict__ Kb,
                 const u16* __restrict__ Vb, u16* __restrict__ Ob) {
    __shared__ u16 sK[32][136];       // [key][d], padded
    __shared__ u16 sVt[128][40];      // [d][key], padded (transposed V)
    __shared__ u16 sP[4][16][40];     // per-wave P tile [m][n]

    const int tid  = threadIdx.x;
    const int lane = tid & 31;
    const int w    = tid >> 5;
    const int r    = lane & 15;
    const int half = lane >> 4;
    const int kb   = half * 8;

    const int qblk = blockIdx.x & 31;
    const int h    = (blockIdx.x >> 5) & 31;
    const int b    = blockIdx.x >> 10;
    const int kvh  = h >> 2;                  // repeat_interleave on heads
    const int q0   = qblk * 64 + w * 16;      // wave q-tile base row

    // Q fragments: 4 chunks of 16x32 along HD
    FragB aq[4];
    {
        const u16* qp = Qb + (((size_t)b * S_C + q0 + r) * NH_C + h) * HD_C;
        #pragma unroll
        for (int c = 0; c < 4; ++c) {
            aq[c].u[0] = *(const uint4*)(qp + c * 32 + kb);
            aq[c].u[1] = *(const uint4*)(qp + c * 32 + 16 + kb);
        }
    }

    f32x8 acc[8] = {};
    float ms[8], ls[8];
    #pragma unroll
    for (int v = 0; v < 8; ++v) { ms[v] = -3.0e38f; ls[v] = 0.0f; }

    const float scale = 0.08838834764831845f;  // 1/sqrt(128)
    const int nblk = qblk * 2 + 2;             // key blocks up to block diag

    // staging assignments (fixed per thread)
    const int kkey  = tid >> 2;                // K: one key, 64B strip
    const int kdseg = (tid & 3) * 32;
    const unsigned laK = lds_addr(&sK[kkey][kdseg]);
    const int vkp   = tid & 15;                // V: key pair 2*vkp, 2*vkp+1
    const int vdseg = (tid >> 4) * 16;         // 16 d values

    for (int blk = 0; blk < nblk; ++blk) {
        const int k0 = blk * 32;
        __syncthreads();
        // ---- K panel: async global -> LDS ----
        async_copy64(laK, (unsigned long long)(
            Kb + (((size_t)b * S_C + k0 + kkey) * NKV_C + kvh) * HD_C + kdseg));
        // ---- V panel: load two keys, store transposed as packed u32 ----
        {
            const u16* vp0 = Vb + (((size_t)b * S_C + k0 + 2 * vkp) * NKV_C + kvh) * HD_C + vdseg;
            const u16* vp1 = vp0 + (size_t)NKV_C * HD_C;
            V16 v0, v1;
            v0.q[0] = ((const uint4*)vp0)[0];  v0.q[1] = ((const uint4*)vp0)[1];
            v1.q[0] = ((const uint4*)vp1)[0];  v1.q[1] = ((const uint4*)vp1)[1];
            #pragma unroll
            for (int e = 0; e < 16; ++e) {
                unsigned pk = (unsigned)v0.h[e] | ((unsigned)v1.h[e] << 16);
                *(unsigned*)&sVt[vdseg + e][2 * vkp] = pk;
            }
        }
        wait_async0();
        __syncthreads();

        // ---- scores: 2 key sub-tiles x 4 HD chunks ----
        f32x8 st[2];
        #pragma unroll
        for (int t = 0; t < 2; ++t) {
            f32x8 s = {};
            #pragma unroll
            for (int c = 0; c < 4; ++c) {
                FragB bk;
                const u16* p = &sK[t * 16 + r][c * 32 + kb];
                bk.u[0] = *(const uint4*)p;
                bk.u[1] = *(const uint4*)(p + 16);
                s = wmma_bf16(aq[c], bk, s);
            }
            st[t] = s;
        }

        // ---- causal mask + online softmax (rows span 16 lanes) ----
        float pm[8];
        #pragma unroll
        for (int v = 0; v < 8; ++v) {
            int qrow = q0 + 8 * half + v;
            float s0 = (k0 + r      <= qrow) ? st[0][v] * scale : -3.0e38f;
            float s1 = (k0 + 16 + r <= qrow) ? st[1][v] * scale : -3.0e38f;
            float x = fmaxf(s0, s1);
            x = fmaxf(x, __shfl_xor(x, 1, 32));
            x = fmaxf(x, __shfl_xor(x, 2, 32));
            x = fmaxf(x, __shfl_xor(x, 4, 32));
            x = fmaxf(x, __shfl_xor(x, 8, 32));
            float nm   = fmaxf(ms[v], x);
            float corr = __expf(ms[v] - nm);
            ms[v] = nm;
            float p0 = __expf(s0 - nm);
            float p1 = __expf(s1 - nm);
            float rs = p0 + p1;
            rs += __shfl_xor(rs, 1, 32);
            rs += __shfl_xor(rs, 2, 32);
            rs += __shfl_xor(rs, 4, 32);
            rs += __shfl_xor(rs, 8, 32);
            ls[v] = ls[v] * corr + rs;
            pm[v] = corr;
            st[0][v] = p0;
            st[1][v] = p1;
        }
        #pragma unroll
        for (int d = 0; d < 8; ++d)
            #pragma unroll
            for (int v = 0; v < 8; ++v)
                acc[d][v] *= pm[v];

        // ---- P: C/D layout -> LDS -> A-fragment layout ----
        #pragma unroll
        for (int v = 0; v < 8; ++v) {
            sP[w][8 * half + v][r]      = f2bf(st[0][v]);
            sP[w][8 * half + v][16 + r] = f2bf(st[1][v]);
        }
        asm volatile("s_wait_dscnt 0x0" ::: "memory");
        FragB ap;
        {
            const u16* p = &sP[w][r][kb];
            ap.u[0] = *(const uint4*)p;
            ap.u[1] = *(const uint4*)(p + 16);
        }

        // ---- O += P @ V : 8 d-tiles ----
        #pragma unroll
        for (int d = 0; d < 8; ++d) {
            FragB bv;
            const u16* p = &sVt[d * 16 + r][kb];
            bv.u[0] = *(const uint4*)p;
            bv.u[1] = *(const uint4*)(p + 16);
            acc[d] = wmma_bf16(ap, bv, acc[d]);
        }
    }

    // ---- epilogue: normalize, write bf16 [B, S, H] ----
    #pragma unroll
    for (int v = 0; v < 8; ++v) {
        float inv = 1.0f / ls[v];
        int qrow = q0 + 8 * half + v;
        u16* op = Ob + ((size_t)b * S_C + qrow) * H_C + h * HD_C;
        #pragma unroll
        for (int d = 0; d < 8; ++d)
            op[d * 16 + r] = f2bf(acc[d][v] * inv);
    }
}

// ---------------------------------------------------------------------------
// Launch
// ---------------------------------------------------------------------------
extern "C" void kernel_launch(void* const* d_in, const int* in_sizes, int n_in,
                              void* d_out, int out_size, void* d_ws, size_t ws_size,
                              hipStream_t stream) {
    (void)in_sizes; (void)n_in; (void)out_size; (void)ws_size;
    const float* hs  = (const float*)d_in[0];
    const int*   pos = (const int*)d_in[1];
    const float* Wq  = (const float*)d_in[2];
    const float* bq  = (const float*)d_in[3];
    const float* Wk  = (const float*)d_in[4];
    const float* bk  = (const float*)d_in[5];
    const float* Wv  = (const float*)d_in[6];
    const float* bv  = (const float*)d_in[7];
    const float* Wo  = (const float*)d_in[8];
    const float* bo  = (const float*)d_in[9];
    float* out = (float*)d_out;

    const long BS  = (long)B_C * S_C;            // 4096
    const long KVD = (long)NKV_C * HD_C;         // 1024

    size_t off = 0;
    char* base = (char*)d_ws;
    auto carve = [&](size_t bytes) -> void* {
        void* p = base + off;
        off += (bytes + 255) & ~(size_t)255;
        return p;
    };
    u16*   Xbf = (u16*)carve((size_t)BS * H_C * 2);
    u16*   Wqb = (u16*)carve((size_t)H_C * H_C * 2);
    u16*   Wkb = (u16*)carve((size_t)KVD * H_C * 2);
    u16*   Wvb = (u16*)carve((size_t)KVD * H_C * 2);
    u16*   Wob = (u16*)carve((size_t)H_C * H_C * 2);
    float* Qf  = (float*)carve((size_t)BS * H_C * 4);
    float* Kf  = (float*)carve((size_t)BS * KVD * 4);
    float* Vf  = (float*)carve((size_t)BS * KVD * 4);
    u16*   Qbf = (u16*)carve((size_t)BS * H_C * 2);
    u16*   Kbf = (u16*)carve((size_t)BS * KVD * 2);
    u16*   Vbf = (u16*)carve((size_t)BS * KVD * 2);
    u16*   Abf = (u16*)carve((size_t)BS * H_C * 2);

    auto cast = [&](const float* src, u16* dst, long n) {
        cast_bf16_kernel<<<(unsigned)((n + 255) / 256), 256, 0, stream>>>(src, dst, n);
    };
    cast(hs, Xbf, BS * H_C);
    cast(Wq, Wqb, (long)H_C * H_C);
    cast(Wk, Wkb, KVD * H_C);
    cast(Wv, Wvb, KVD * H_C);
    cast(Wo, Wob, (long)H_C * H_C);

    // Projections (tile 128 x 256)
    gemm_bf16_kernel<<<dim3(BS / 128, H_C / 256), 256, 0, stream>>>(
        Xbf, Wqb, bq, Qf, (int)BS, H_C, H_C);
    gemm_bf16_kernel<<<dim3(BS / 128, (int)KVD / 256), 256, 0, stream>>>(
        Xbf, Wkb, bk, Kf, (int)BS, (int)KVD, H_C);
    gemm_bf16_kernel<<<dim3(BS / 128, (int)KVD / 256), 256, 0, stream>>>(
        Xbf, Wvb, bv, Vf, (int)BS, (int)KVD, H_C);

    // RoPE + cast
    {
        long qp = BS * NH_C * (HD_C / 2);
        rope_cast_kernel<<<(unsigned)((qp + 255) / 256), 256, 0, stream>>>(
            Qf, pos, Qbf, NH_C, qp);
        long kp = BS * NKV_C * (HD_C / 2);
        rope_cast_kernel<<<(unsigned)((kp + 255) / 256), 256, 0, stream>>>(
            Kf, pos, Kbf, NKV_C, kp);
        cast(Vf, Vbf, BS * KVD);
    }

    // Attention: B * NH * (S/64) blocks of 4 waves
    attn_kernel<<<B_C * NH_C * (S_C / 64), 128, 0, stream>>>(Qbf, Kbf, Vbf, Abf);

    // Output projection -> f32 out
    gemm_bf16_kernel<<<dim3(BS / 128, H_C / 256), 256, 0, stream>>>(
        Abf, Wob, bo, out, (int)BS, H_C, H_C);
}
